// ProtoNet_9320079032854
// MI455X (gfx1250) — matrix operand order repaired and checked by reading, not compiled, not run on hardware
//
#include <hip/hip_runtime.h>
#include <hip/hip_bf16.h>

// ---------------------------------------------------------------------------
// Types for CDNA5 WMMA (wave32): A/B = 16 halfs per lane, C/D = 8 floats/lane
// ---------------------------------------------------------------------------
typedef __attribute__((ext_vector_type(16))) _Float16 v16h;
typedef __attribute__((ext_vector_type(8)))  _Float16 v8h;
typedef __attribute__((ext_vector_type(8)))  float    v8f;

__device__ __forceinline__ v8f wmma_f16(v16h a, v16h b, v8f c) {
  // D = A(16x32) * B(32x16) + C, f32 accumulate
  return __builtin_amdgcn_wmma_f32_16x16x32_f16(false, a, false, b, (short)0, c,
                                                false, false);
}

// A fragment: 16x32 f16, src row-major with leading dim ld (in halfs).
// Lane l: row = l&15; lanes 0-15 hold K {0..7,16..23}, lanes 16-31 {8..15,24..31}.
__device__ __forceinline__ v16h load_a_frag(const _Float16* __restrict__ src, int ld) {
  int lane = threadIdx.x & 31;
  int row  = lane & 15;
  int kofs = (lane & 16) ? 8 : 0;
  const _Float16* p = src + (size_t)row * ld + kofs;
  v8h lo = *(const v8h*)(p);
  v8h hi = *(const v8h*)(p + 16);
  v16h a;
#pragma unroll
  for (int i = 0; i < 8; ++i) { a[i] = lo[i]; a[i + 8] = hi[i]; }
  return a;
}

// B fragment: B[k][n] = src[n*ld + k]  (i.e. B = W^T where W is row-major [N,K]).
// Lane l: col = l&15; lanes 0-15 hold K {0..7,16..23}, lanes 16-31 {8..15,24..31}.
__device__ __forceinline__ v16h load_b_frag(const _Float16* __restrict__ src, int ld) {
  int lane = threadIdx.x & 31;
  int col  = lane & 15;
  int kofs = (lane & 16) ? 8 : 0;
  const _Float16* p = src + (size_t)col * ld + kofs;
  v8h lo = *(const v8h*)(p);
  v8h hi = *(const v8h*)(p + 16);
  v16h b;
#pragma unroll
  for (int i = 0; i < 8; ++i) { b[i] = lo[i]; b[i + 8] = hi[i]; }
  return b;
}

// ---------------------------------------------------------------------------
// f32 -> f16 conversion (weights)
// ---------------------------------------------------------------------------
__global__ void convert_f32_f16_kernel(const float* __restrict__ src,
                                       _Float16* __restrict__ dst, int n) {
  for (int i = blockIdx.x * blockDim.x + threadIdx.x; i < n;
       i += gridDim.x * blockDim.x)
    dst[i] = (_Float16)src[i];
}

// ---------------------------------------------------------------------------
// Embedding gather: 65 sequences x 1024 tokens x 128 dims -> f32 + f16
// ---------------------------------------------------------------------------
__global__ void embed_kernel(const int* __restrict__ sup,
                             const int* __restrict__ qry,
                             const float* __restrict__ emb,
                             float* __restrict__ X, _Float16* __restrict__ Xh) {
  int idx = blockIdx.x * blockDim.x + threadIdx.x;  // exactly 65*1024*128
  int t = idx >> 7, d = idx & 127;
  int tok = (t < 65536) ? sup[t] : qry[t - 65536];
  float v = emb[tok * 128 + d];
  X[idx]  = v;
  Xh[idx] = (_Float16)v;
}

// ---------------------------------------------------------------------------
// Generic GEMM: out = act(A[M,K] * W[N,K]^T + bias).
// One 16x64 strip per wave: one A fragment feeds 4 WMMAs per k-step.
// Requires N % 64 == 0.
// ---------------------------------------------------------------------------
__global__ void gemm_bias_kernel(const _Float16* __restrict__ A,
                                 const _Float16* __restrict__ W,
                                 const float* __restrict__ bias,
                                 _Float16* __restrict__ outh,
                                 float* __restrict__ outf,
                                 int M, int N, int K, int relu) {
  int wave = (blockIdx.x * blockDim.x + threadIdx.x) >> 5;
  int mt = M >> 4, nt = N >> 6;
  if (wave >= mt * nt) return;  // wave-uniform; EXEC stays all-ones inside
  int tm = wave % mt, tn = wave / mt;

  const _Float16* Ab = A + (size_t)(tm * 16) * K;
  const _Float16* Wb = W + (size_t)(tn * 64) * K;
  v8f acc[4];
#pragma unroll
  for (int j = 0; j < 4; ++j)
#pragma unroll
    for (int i = 0; i < 8; ++i) acc[j][i] = 0.f;

  for (int k0 = 0; k0 < K; k0 += 32) {
    v16h a  = load_a_frag(Ab + k0, K);
    v16h b0 = load_b_frag(Wb + k0, K);
    v16h b1 = load_b_frag(Wb + (size_t)16 * K + k0, K);
    v16h b2 = load_b_frag(Wb + (size_t)32 * K + k0, K);
    v16h b3 = load_b_frag(Wb + (size_t)48 * K + k0, K);
    acc[0] = wmma_f16(a, b0, acc[0]);
    acc[1] = wmma_f16(a, b1, acc[1]);
    acc[2] = wmma_f16(a, b2, acc[2]);
    acc[3] = wmma_f16(a, b3, acc[3]);
  }

  int lane = threadIdx.x & 31;
  int col = lane & 15, rbase = (lane & 16) ? 8 : 0;
#pragma unroll
  for (int j = 0; j < 4; ++j) {
    int ncol = tn * 64 + j * 16 + col;
    float bv = bias ? bias[ncol] : 0.f;
#pragma unroll
    for (int r = 0; r < 8; ++r) {
      float v = acc[j][r] + bv;
      if (relu) v = fmaxf(v, 0.f);
      size_t idx = (size_t)(tm * 16 + rbase + r) * N + ncol;
      if (outh) outh[idx] = (_Float16)v;
      if (outf) outf[idx] = v;
    }
  }
}

// ---------------------------------------------------------------------------
// Fused GEMM (N=128) + bias + residual + LayerNorm. Block = 8 waves = 16 rows.
// ---------------------------------------------------------------------------
__global__ void gemm_res_ln_kernel(const _Float16* __restrict__ A,
                                   const _Float16* __restrict__ W,
                                   const float* __restrict__ bias,
                                   float* __restrict__ X, _Float16* __restrict__ Xh,
                                   const float* __restrict__ ln_s,
                                   const float* __restrict__ ln_b, int K) {
  __shared__ float smem[16][132];
  int wid = threadIdx.x >> 5;   // n-tile 0..7
  int tm  = blockIdx.x;         // 16-row tile

  const _Float16* Ab = A + (size_t)(tm * 16) * K;
  const _Float16* Wb = W + (size_t)(wid * 16) * K;
  v8f c;
#pragma unroll
  for (int i = 0; i < 8; ++i) c[i] = 0.f;
  for (int k0 = 0; k0 < K; k0 += 32) {
    v16h a = load_a_frag(Ab + k0, K);
    v16h b = load_b_frag(Wb + k0, K);
    c = wmma_f16(a, b, c);
  }
  int lane = threadIdx.x & 31;
  int col = lane & 15, rbase = (lane & 16) ? 8 : 0;
  float bv = bias[wid * 16 + col];
#pragma unroll
  for (int r = 0; r < 8; ++r) smem[rbase + r][wid * 16 + col] = c[r] + bv;
  __syncthreads();

  // LayerNorm: 16 threads per row, 8 cols each
  int row = threadIdx.x >> 4, g = threadIdx.x & 15;
  size_t rowoff = (size_t)(tm * 16 + row) * 128;
  float vals[8];
  float s = 0.f, ss = 0.f;
#pragma unroll
  for (int j = 0; j < 8; ++j) {
    int cc = g * 8 + j;
    float v = X[rowoff + cc] + smem[row][cc];  // residual + gemm result
    vals[j] = v; s += v; ss += v * v;
  }
  s  += __shfl_xor(s, 1);  s  += __shfl_xor(s, 2);  s  += __shfl_xor(s, 4);  s  += __shfl_xor(s, 8);
  ss += __shfl_xor(ss, 1); ss += __shfl_xor(ss, 2); ss += __shfl_xor(ss, 4); ss += __shfl_xor(ss, 8);
  float mean = s * (1.f / 128.f);
  float var  = ss * (1.f / 128.f) - mean * mean;
  float inv  = rsqrtf(var + 1e-5f);
#pragma unroll
  for (int j = 0; j < 8; ++j) {
    int cc = g * 8 + j;
    float y = (vals[j] - mean) * inv * ln_s[cc] + ln_b[cc];
    X[rowoff + cc]  = y;
    Xh[rowoff + cc] = (_Float16)y;
  }
}

// ---------------------------------------------------------------------------
// Flash attention: one wave per (b,h,16-row q-tile); the 4 waves of a block
// share one (b,h), so each 32x32 V tile is staged (transposed) into LDS once
// per block with coalesced 16B global loads. P tiles round-trip through LDS
// to convert WMMA C-layout -> A-fragment layout. DH=32, S=1024, no mask.
// ---------------------------------------------------------------------------
__global__ void attn_kernel(const _Float16* __restrict__ QKV,
                            _Float16* __restrict__ O) {
  __shared__ __align__(16) _Float16 pstage[4][16][40];  // per-wave P tile
  __shared__ __align__(16) _Float16 vt[32][40];         // vt[dh][k] = V[k][dh]
  int wid  = threadIdx.x >> 5;
  int gw   = blockIdx.x * 4 + wid;     // 0 .. 65*4*64-1
  int qt   = gw & 63;
  int h    = (gw >> 6) & 3;
  int b    = gw >> 8;
  int lane = threadIdx.x & 31;
  int col = lane & 15, rbase = (lane & 16) ? 8 : 0;

  const _Float16* base = QKV + (size_t)b * 1024 * 384 + h * 32;
  const _Float16* Qb = base + (size_t)(qt * 16) * 384;
  const _Float16* Kb = base + 128;
  const _Float16* Vb = base + 256;

  // cooperative V staging: thread -> (row vr of the 32-row tile, 8-half chunk)
  int vr = threadIdx.x >> 2;
  int vc = (threadIdx.x & 3) * 8;

  v16h aq = load_a_frag(Qb, 384);  // Q tile, loaded once (DH=32 = one A frag)
  v8f o0, o1;
  float mrow[8], lrow[8];
#pragma unroll
  for (int r = 0; r < 8; ++r) { o0[r] = 0.f; o1[r] = 0.f; mrow[r] = -1e30f; lrow[r] = 0.f; }
  const float scale = 0.17677669529663687f;  // 1/sqrt(32)

  for (int kt = 0; kt < 1024; kt += 32) {
    // stage V rows kt..kt+31, transposed, into LDS (each element written once)
    v8h vrow = *(const v8h*)(Vb + (size_t)(kt + vr) * 384 + vc);
#pragma unroll
    for (int i = 0; i < 8; ++i) vt[vc + i][vr] = vrow[i];

    v16h bk0 = load_b_frag(Kb + (size_t)kt * 384, 384);
    v16h bk1 = load_b_frag(Kb + (size_t)(kt + 16) * 384, 384);
    v8f z;
#pragma unroll
    for (int i = 0; i < 8; ++i) z[i] = 0.f;
    v8f s0 = wmma_f16(aq, bk0, z);
    v8f s1 = wmma_f16(aq, bk1, z);

#pragma unroll
    for (int r = 0; r < 8; ++r) {
      float a0 = s0[r] * scale;
      float a1 = s1[r] * scale;
      float v = fmaxf(a0, a1);
      v = fmaxf(v, __shfl_xor(v, 1)); v = fmaxf(v, __shfl_xor(v, 2));
      v = fmaxf(v, __shfl_xor(v, 4)); v = fmaxf(v, __shfl_xor(v, 8));
      float mn = fmaxf(mrow[r], v);
      float alpha = __expf(mrow[r] - mn);
      float p0 = __expf(a0 - mn);
      float p1 = __expf(a1 - mn);
      float rs = p0 + p1;
      rs += __shfl_xor(rs, 1); rs += __shfl_xor(rs, 2);
      rs += __shfl_xor(rs, 4); rs += __shfl_xor(rs, 8);
      lrow[r] = lrow[r] * alpha + rs;
      mrow[r] = mn;
      o0[r] *= alpha;
      o1[r] *= alpha;
      pstage[wid][rbase + r][col]      = (_Float16)p0;
      pstage[wid][rbase + r][col + 16] = (_Float16)p1;
    }
    __syncthreads();
    // P (A-frag) from LDS; V (B-frags) from transposed LDS tile: contiguous 16B reads
    v16h ap  = load_a_frag(&pstage[wid][0][0], 40);
    v16h bv0 = load_b_frag(&vt[0][0], 40);   // dh cols 0..15
    v16h bv1 = load_b_frag(&vt[16][0], 40);  // dh cols 16..31
    o0 = wmma_f16(ap, bv0, o0);
    o1 = wmma_f16(ap, bv1, o1);
    __syncthreads();
  }

#pragma unroll
  for (int r = 0; r < 8; ++r) {
    float inv = 1.f / lrow[r];
    int row = qt * 16 + rbase + r;
    size_t off = ((size_t)b * 1024 + row) * 128 + h * 32;
    O[off + col]      = (_Float16)(o0[r] * inv);
    O[off + 16 + col] = (_Float16)(o1[r] * inv);
  }
}

// ---------------------------------------------------------------------------
// Prototype reduction, stage 1: 64 blocks x 1024 tokens, per-class partials.
// Deterministic (fixed iteration order, no atomics).
// ---------------------------------------------------------------------------
__global__ void proto_partial_kernel(const float* __restrict__ feat,
                                     const int* __restrict__ outs,
                                     const int* __restrict__ masks,
                                     float* __restrict__ psums,   // [64][10][256]
                                     float* __restrict__ pcnts) { // [64][10]
  int blk = blockIdx.x, d = threadIdx.x;
  float s[10], cnt[10];
#pragma unroll
  for (int c = 0; c < 10; ++c) { s[c] = 0.f; cnt[c] = 0.f; }
  int t0 = blk * 1024;
  for (int i = 0; i < 1024; ++i) {
    int t = t0 + i;
    int o = outs[t];
    bool ok = (masks[t] > 0) && (o < 10);
    float fv = ok ? feat[(size_t)t * 256 + d] : 0.f;
#pragma unroll
    for (int c = 0; c < 10; ++c) {
      bool hit = ok && (o == c);
      s[c]   += hit ? fv : 0.f;
      cnt[c] += hit ? 1.f : 0.f;
    }
  }
#pragma unroll
  for (int c = 0; c < 10; ++c)
    psums[((size_t)blk * 10 + c) * 256 + d] = s[c];
  if (d == 0) {
#pragma unroll
    for (int c = 0; c < 10; ++c) pcnts[blk * 10 + c] = cnt[c];
  }
}

// Stage 2: fixed-order 64-way combine -> prototypes [10][256]
__global__ void proto_final_kernel(const float* __restrict__ psums,
                                   const float* __restrict__ pcnts,
                                   float* __restrict__ protos) {
  int c = blockIdx.x, d = threadIdx.x;
  float s = 0.f, cnt = 0.f;
  for (int blk = 0; blk < 64; ++blk) {
    s   += psums[((size_t)blk * 10 + c) * 256 + d];
    cnt += pcnts[blk * 10 + c];
  }
  protos[c * 256 + d] = (cnt > 0.f) ? s / cnt : 0.f;
}

// ---------------------------------------------------------------------------
// Logits: -||qf - proto||^2, [1024, 10]
// ---------------------------------------------------------------------------
__global__ void logits_kernel(const float* __restrict__ feat,
                              const float* __restrict__ protos,
                              float* __restrict__ out) {
  int idx = blockIdx.x * blockDim.x + threadIdx.x;  // 0..10239
  int s = idx / 10, c = idx % 10;
  const float* q = feat + ((size_t)65536 + s) * 256;
  const float* p = protos + c * 256;
  float acc = 0.f;
  for (int d = 0; d < 256; ++d) {
    float df = q[d] - p[d];
    acc += df * df;
  }
  out[idx] = -acc;
}

// ---------------------------------------------------------------------------
// Host-side orchestration
// ---------------------------------------------------------------------------
extern "C" void kernel_launch(void* const* d_in, const int* in_sizes, int n_in,
                              void* d_out, int out_size, void* d_ws, size_t ws_size,
                              hipStream_t stream) {
  (void)in_sizes; (void)n_in; (void)out_size; (void)ws_size;
  const int*   sup_in  = (const int*)d_in[0];
  const int*   sup_out = (const int*)d_in[1];
  const int*   qry     = (const int*)d_in[2];
  const int*   masks   = (const int*)d_in[3];
  const float* emb     = (const float*)d_in[4];
  const float* Wqkv    = (const float*)d_in[5];
  const float* bqkv    = (const float*)d_in[6];
  const float* Wo      = (const float*)d_in[7];
  const float* bo      = (const float*)d_in[8];
  const float* ln1_s   = (const float*)d_in[9];
  const float* ln1_b   = (const float*)d_in[10];
  const float* W1      = (const float*)d_in[11];
  const float* b1      = (const float*)d_in[12];
  const float* W2      = (const float*)d_in[13];
  const float* b2      = (const float*)d_in[14];
  const float* ln2_s   = (const float*)d_in[15];
  const float* ln2_b   = (const float*)d_in[16];
  const float* Wp      = (const float*)d_in[17];
  const float* bp      = (const float*)d_in[18];

  const int M = 65 * 1024;  // 64 support seqs + 1 query seq, S=1024
  size_t off = 0;
  auto take = [&](size_t bytes) -> void* {
    void* p = (char*)d_ws + off;
    off += (bytes + 255) & ~(size_t)255;
    return p;
  };
  _Float16* WqkvH = (_Float16*)take((size_t)3 * 384 * 128 * 2);
  _Float16* WoH   = (_Float16*)take((size_t)3 * 128 * 128 * 2);
  _Float16* W1H   = (_Float16*)take((size_t)3 * 256 * 128 * 2);
  _Float16* W2H   = (_Float16*)take((size_t)3 * 128 * 256 * 2);
  _Float16* WpH   = (_Float16*)take((size_t)256 * 128 * 2);
  float*    X     = (float*)take((size_t)M * 128 * 4);
  _Float16* Xh    = (_Float16*)take((size_t)M * 128 * 2);
  _Float16* Oh    = (_Float16*)take((size_t)M * 128 * 2);
  void*     big   = take((size_t)M * 256 * 4);  // QKVh (M*384 f16) / Fh (M*256 f16) / feat (M*256 f32)
  float*    protos = (float*)take((size_t)10 * 256 * 4);
  float*    psums  = (float*)take((size_t)64 * 10 * 256 * 4);
  float*    pcnts  = (float*)take((size_t)64 * 10 * 4);

  _Float16* QKVh = (_Float16*)big;
  _Float16* Fh   = (_Float16*)big;
  float*    feat = (float*)big;

  // Weight conversion f32 -> f16
  convert_f32_f16_kernel<<<512, 256, 0, stream>>>(Wqkv, WqkvH, 3 * 384 * 128);
  convert_f32_f16_kernel<<<192, 256, 0, stream>>>(Wo,   WoH,   3 * 128 * 128);
  convert_f32_f16_kernel<<<384, 256, 0, stream>>>(W1,   W1H,   3 * 256 * 128);
  convert_f32_f16_kernel<<<384, 256, 0, stream>>>(W2,   W2H,   3 * 128 * 256);
  convert_f32_f16_kernel<<<128, 256, 0, stream>>>(Wp,   WpH,   256 * 128);

  // Embedding: 65*1024*128 elements
  embed_kernel<<<33280, 256, 0, stream>>>(sup_in, qry, emb, X, Xh);

  const int mt = M / 16;  // 4160
  for (int l = 0; l < 3; ++l) {
    const _Float16* WqkvL = WqkvH + (size_t)l * 384 * 128;
    const _Float16* WoL   = WoH   + (size_t)l * 128 * 128;
    const _Float16* W1L   = W1H   + (size_t)l * 256 * 128;
    const _Float16* W2L   = W2H   + (size_t)l * 128 * 256;

    // QKV = Xh * Wqkv^T + bqkv   -> [M, 384] f16   (waves = mt * 384/64 = mt*6)
    gemm_bias_kernel<<<(mt * 6) / 4, 128, 0, stream>>>(
        Xh, WqkvL, bqkv + l * 384, QKVh, (float*)nullptr, M, 384, 128, 0);

    // Flash attention -> Oh [M,128] f16
    attn_kernel<<<65 * 4 * 64 / 4, 128, 0, stream>>>(QKVh, Oh);

    // X = LN(X + Oh * Wo^T + bo)
    gemm_res_ln_kernel<<<mt, 256, 0, stream>>>(
        Oh, WoL, bo + l * 128, X, Xh, ln1_s + l * 128, ln1_b + l * 128, 128);

    // F = relu(Xh * W1^T + b1)  -> [M,256] f16   (waves = mt * 256/64 = mt*4)
    gemm_bias_kernel<<<mt, 128, 0, stream>>>(
        Xh, W1L, b1 + l * 256, Fh, (float*)nullptr, M, 256, 128, 1);

    // X = LN(X + F * W2^T + b2)
    gemm_res_ln_kernel<<<mt, 256, 0, stream>>>(
        Fh, W2L, b2 + l * 128, X, Xh, ln2_s + l * 128, ln2_b + l * 128, 256);
  }

  // feat = Xh * Wp^T + bp  -> [M,256] f32
  gemm_bias_kernel<<<mt, 128, 0, stream>>>(
      Xh, WpH, bp, (_Float16*)nullptr, feat, M, 256, 128, 0);

  // Prototypes (two-stage deterministic reduction) and logits
  proto_partial_kernel<<<64, 256, 0, stream>>>(feat, sup_out, masks, psums, pcnts);
  proto_final_kernel<<<10, 256, 0, stream>>>(psums, pcnts, protos);
  logits_kernel<<<40, 256, 0, stream>>>(feat, protos, (float*)d_out);
}
